// Model_32787780338333
// MI455X (gfx1250) — compile-verified, matrix operand order checked by reading, not run on hardware
//
#include <hip/hip_runtime.h>
#include <hip/hip_bf16.h>

typedef __attribute__((ext_vector_type(16))) _Float16 v16h;
typedef __attribute__((ext_vector_type(8)))  float    v8f;
typedef __attribute__((ext_vector_type(4)))  int      v4i;

#define DM   512      // d_model
#define NG   2048     // 4*d_model (gates)
#define BB   128      // batch
#define TT   128      // seq len
#define DFF  2048
#define KIN  1056     // 32 + 512 + 512
#define MROWS (BB*TT) // 16384 rows per stream

// ---------------------------------------------------------------------------
// WMMA fragment layout (ISA 7.12.2, 16-bit A 16x32 / B 32x16):
//   lane = (row|col & 15) + ((kk & 8) ? 16 : 0)
//   slot = (kk & 7) + ((kk & 16) >> 1)
// Operands pre-swizzled so each lane's 16 halves are CONTIGUOUS (32 B):
// fragment load = 2x ds_load_b128. A row-major 8-half group (r, kk8*8..+7)
// lands contiguously at lane = (r&15) + ((kk8&1)<<4), slotbase = (kk8>>1)*8.
// ---------------------------------------------------------------------------

// -------- async global->LDS copy (GLOBAL_LOAD_ASYNC_TO_LDS_B128, ASYNCcnt) --------
#if defined(__gfx1250__) && __has_builtin(__builtin_amdgcn_global_load_async_to_lds_b128) && \
    __has_builtin(__builtin_amdgcn_s_wait_asynccnt)
#define ASYNC_CP 1
#else
#define ASYNC_CP 0
#endif

#if ASYNC_CP
typedef __attribute__((address_space(1))) v4i* gptr_v4i;
typedef __attribute__((address_space(3))) v4i* lptr_v4i;
#endif

__device__ __forceinline__ void cp_b128(void* lds_dst, const void* gsrc) {
#if ASYNC_CP
    __builtin_amdgcn_global_load_async_to_lds_b128((gptr_v4i)gsrc, (lptr_v4i)lds_dst,
                                                   0, 0);
#else
    *(uint4*)lds_dst = *(const uint4*)gsrc;
#endif
}
__device__ __forceinline__ void cp_wait() {
#if ASYNC_CP
    __builtin_amdgcn_s_wait_asynccnt(0);
#endif
}

__device__ __forceinline__ v8f wmma_f16(v16h a, v16h b, v8f c) {
    return __builtin_amdgcn_wmma_f32_16x16x32_f16(false, a, false, b, (short)0, c,
                                                  false, false);
}
__device__ __forceinline__ float sigmoidf_(float x) { return 1.f / (1.f + __expf(-x)); }

union Pack8 { uint4 u; _Float16 h[8]; };

// ---------------- LSTM step: gates GEMM (fused [X|H] @ Wg) + cell update ----------
// grid.x = 32 (16 h-columns j0, all 4 gate sections); block 256 (wave w: rows 16w..)
__global__ __launch_bounds__(256)
void lstm_step(const float* __restrict__ xsrc, int x_row_stride, int x_off, int in_dim,
               const float* __restrict__ hin, const float* __restrict__ cin,
               const _Float16* __restrict__ Wsw,  // swizzled [(KC,128 tiles,32,16)]
               const float* __restrict__ bias,    // [2048] = b_ih + b_hh
               float* __restrict__ hout, float* __restrict__ cout,
               _Float16* __restrict__ wh16, _Float16* __restrict__ wc16, int lag_t)
{
    __shared__ alignas(16) _Float16 ldsA[8 * 512];   // 8 m-tiles, frag order
    __shared__ alignas(16) _Float16 ldsB[4 * 512];   // 4 gate tiles, frag order
    const int tid = threadIdx.x;
    const int wave = tid >> 5, lane = tid & 31;
    const int j0 = blockIdx.x * 16;
    const int nt_j = j0 >> 4;
    const int KC = (in_dim + DM) >> 5;

    v8f acc[4];
#pragma unroll
    for (int g = 0; g < 4; ++g)
#pragma unroll
        for (int e = 0; e < 8; ++e) acc[g][e] = 0.f;

    for (int kc = 0; kc < KC; ++kc) {
        // stage B first: 4 gate tiles (pre-swizzled global) -> async b128, 1/thread
        {
            int g = tid >> 6, idx = tid & 63;
            const uint4* src =
                (const uint4*)(Wsw + ((size_t)kc * (NG >> 4) + nt_j + g * 32) * 512);
            cp_b128(((uint4*)ldsB) + g * 64 + idx, src + idx);
        }
        // stage A: 128 rows x 4 groups-of-8 = 512 groups, 2 per thread, packed b128
        for (int i = tid; i < 512; i += 256) {
            int r = i >> 2, g8 = i & 3;
            int kg = kc * 32 + g8 * 8;
            Pack8 p;
            const float* s = (kg < in_dim)
                                 ? (xsrc + (size_t)r * x_row_stride + x_off + kg)
                                 : (hin + (size_t)r * DM + (kg - in_dim));
#pragma unroll
            for (int e = 0; e < 8; ++e) p.h[e] = (_Float16)s[e];
            int laneT = (r & 15) + ((g8 & 1) << 4);
            *(uint4*)&ldsA[(((r >> 4) * 32 + laneT) << 4) + ((g8 >> 1) << 3)] = p.u;
        }
        cp_wait();
        __syncthreads();
        // preload all fragments, then WMMA chain (one dscnt wait, 4 back-to-back wmma)
        v16h a  = *(const v16h*)&ldsA[((wave * 32 + lane) << 4)];
        v16h b0 = *(const v16h*)&ldsB[((0 * 32 + lane) << 4)];
        v16h b1 = *(const v16h*)&ldsB[((1 * 32 + lane) << 4)];
        v16h b2 = *(const v16h*)&ldsB[((2 * 32 + lane) << 4)];
        v16h b3 = *(const v16h*)&ldsB[((3 * 32 + lane) << 4)];
        acc[0] = wmma_f16(a, b0, acc[0]);
        acc[1] = wmma_f16(a, b1, acc[1]);
        acc[2] = wmma_f16(a, b2, acc[2]);
        acc[3] = wmma_f16(a, b3, acc[3]);
        __syncthreads();
    }

    // epilogue: LSTM cell nonlinearity (lane: col j, 8 rows)
    const int nl = lane & 15, half = lane >> 4;
    const int j = j0 + nl;
#pragma unroll
    for (int v = 0; v < 8; ++v) {
        int m = wave * 16 + v + half * 8;
        float gi = acc[0][v] + bias[0 * DM + j];
        float gf = acc[1][v] + bias[1 * DM + j];
        float gg = acc[2][v] + bias[2 * DM + j];
        float go = acc[3][v] + bias[3 * DM + j];
        float cn = sigmoidf_(gf) * cin[m * DM + j] + sigmoidf_(gi) * tanhf(gg);
        float hn = sigmoidf_(go) * tanhf(cn);
        cout[m * DM + j] = cn;
        hout[m * DM + j] = hn;
        if (wh16 != nullptr && lag_t < TT) {   // lagged state for the MLP
            size_t o = ((size_t)m * TT + lag_t) * DM + j;
            wh16[o] = (_Float16)hn;
            wc16[o] = (_Float16)cn;
        }
    }
}

// ---------------- MLP GEMMs: block tile 64x256, 8 waves (each 32x64), dbl-buffer ---
// wave: mt = wave&1 (rows mt*32..+31), ng = wave>>1 (cols ng*64..+63)
__global__ __launch_bounds__(256)
void mlp_gemm0(const float* __restrict__ ysrc, const _Float16* __restrict__ wh16,
               const _Float16* __restrict__ wc16, const _Float16* __restrict__ Wsw,
               const float* __restrict__ bias, _Float16* __restrict__ Z)
{
    __shared__ alignas(16) _Float16 ldsA[2][4 * 512];
    __shared__ alignas(16) _Float16 ldsB[2][16 * 512];
    const int tid = threadIdx.x, wave = tid >> 5, lane = tid & 31;
    const int mt = wave & 1, ng = wave >> 1;
    const int m0 = blockIdx.x * 64;
    const int n0 = blockIdx.y * 256;
    const int nt0 = n0 >> 4;
    const int KC = KIN >> 5;   // 33

    v8f acc[2][4];
#pragma unroll
    for (int s = 0; s < 2; ++s)
#pragma unroll
        for (int t = 0; t < 4; ++t)
#pragma unroll
            for (int e = 0; e < 8; ++e) acc[s][t][e] = 0.f;

    auto stage = [&](int buf, int kc) {
        { // B: 16 contiguous pre-swizzled tiles = 1024 uint4, async, 4 per thread
            const uint4* src = (const uint4*)(Wsw + ((size_t)kc * (DFF >> 4) + nt0) * 512);
            uint4* dst = (uint4*)&ldsB[buf][0];
#pragma unroll
            for (int i = 0; i < 4; ++i) cp_b128(dst + tid + i * 256, src + tid + i * 256);
        }
        { // A: 64 rows x 4 groups = 256 groups, one per thread (gather + cvt path)
            int r = tid >> 2, g8 = tid & 3;
            int m = m0 + r;
            int kg = kc * 32 + g8 * 8;
            int laneT = (r & 15) + ((g8 & 1) << 4);
            _Float16* dst =
                &ldsA[buf][(((r >> 4) * 32 + laneT) << 4) + ((g8 >> 1) << 3)];
            if (kg < 32) {
                Pack8 p;
                const float* s = ysrc + (size_t)m * 32 + kg;
#pragma unroll
                for (int e = 0; e < 8; ++e) p.h[e] = (_Float16)s[e];
                *(uint4*)dst = p.u;
            } else {
                const _Float16* s = (kg < 544) ? (wh16 + (size_t)m * DM + (kg - 32))
                                               : (wc16 + (size_t)m * DM + (kg - 544));
                *(uint4*)dst = *(const uint4*)s;
            }
        }
    };
    auto compute = [&](int buf) {
        v16h a0 = *(const v16h*)&ldsA[buf][(((mt * 2 + 0) * 32 + lane) << 4)];
        v16h a1 = *(const v16h*)&ldsA[buf][(((mt * 2 + 1) * 32 + lane) << 4)];
        v16h b0 = *(const v16h*)&ldsB[buf][(((ng * 4 + 0) * 32 + lane) << 4)];
        v16h b1 = *(const v16h*)&ldsB[buf][(((ng * 4 + 1) * 32 + lane) << 4)];
        v16h b2 = *(const v16h*)&ldsB[buf][(((ng * 4 + 2) * 32 + lane) << 4)];
        v16h b3 = *(const v16h*)&ldsB[buf][(((ng * 4 + 3) * 32 + lane) << 4)];
        acc[0][0] = wmma_f16(a0, b0, acc[0][0]);
        acc[1][0] = wmma_f16(a1, b0, acc[1][0]);
        acc[0][1] = wmma_f16(a0, b1, acc[0][1]);
        acc[1][1] = wmma_f16(a1, b1, acc[1][1]);
        acc[0][2] = wmma_f16(a0, b2, acc[0][2]);
        acc[1][2] = wmma_f16(a1, b2, acc[1][2]);
        acc[0][3] = wmma_f16(a0, b3, acc[0][3]);
        acc[1][3] = wmma_f16(a1, b3, acc[1][3]);
    };

    stage(0, 0);
    cp_wait();
    __syncthreads();
    for (int kc = 0; kc < KC; ++kc) {
        int cur = kc & 1;
        if (kc + 1 < KC) stage(cur ^ 1, kc + 1);   // async B overlaps compute below
        compute(cur);
        cp_wait();
        __syncthreads();
    }

    const int nl = lane & 15, half = lane >> 4;
#pragma unroll
    for (int sm = 0; sm < 2; ++sm)
#pragma unroll
        for (int nt = 0; nt < 4; ++nt) {
            int n = n0 + (ng * 4 + nt) * 16 + nl;
            float bb = bias[n];
#pragma unroll
            for (int v = 0; v < 8; ++v) {
                int m = m0 + (mt * 2 + sm) * 16 + v + half * 8;
                float z = acc[sm][nt][v] + bb;
                z = z > 0.f ? z : expm1f(z);   // ELU
                Z[(size_t)m * DFF + n] = (_Float16)z;
            }
        }
}

__global__ __launch_bounds__(256)
void mlp_gemm1(const _Float16* __restrict__ Zin, const _Float16* __restrict__ Wsw,
               const float* __restrict__ bias, _Float16* __restrict__ Z)
{
    __shared__ alignas(16) _Float16 ldsA[2][4 * 512];
    __shared__ alignas(16) _Float16 ldsB[2][16 * 512];
    const int tid = threadIdx.x, wave = tid >> 5, lane = tid & 31;
    const int mt = wave & 1, ng = wave >> 1;
    const int m0 = blockIdx.x * 64;
    const int n0 = blockIdx.y * 256;
    const int nt0 = n0 >> 4;
    const int KC = DFF >> 5;   // 64

    v8f acc[2][4];
#pragma unroll
    for (int s = 0; s < 2; ++s)
#pragma unroll
        for (int t = 0; t < 4; ++t)
#pragma unroll
            for (int e = 0; e < 8; ++e) acc[s][t][e] = 0.f;

    auto stage = [&](int buf, int kc) {
        { // B: 1024 async b128, 4 per thread
            const uint4* src = (const uint4*)(Wsw + ((size_t)kc * (DFF >> 4) + nt0) * 512);
            uint4* dst = (uint4*)&ldsB[buf][0];
#pragma unroll
            for (int i = 0; i < 4; ++i) cp_b128(dst + tid + i * 256, src + tid + i * 256);
        }
        { // A direct f16: async b128 into swizzled slot, one per thread
            int r = tid >> 2, g8 = tid & 3;
            int laneT = (r & 15) + ((g8 & 1) << 4);
            const uint4* s =
                (const uint4*)(Zin + (size_t)(m0 + r) * DFF + kc * 32 + g8 * 8);
            cp_b128(&ldsA[buf][(((r >> 4) * 32 + laneT) << 4) + ((g8 >> 1) << 3)], s);
        }
    };
    auto compute = [&](int buf) {
        v16h a0 = *(const v16h*)&ldsA[buf][(((mt * 2 + 0) * 32 + lane) << 4)];
        v16h a1 = *(const v16h*)&ldsA[buf][(((mt * 2 + 1) * 32 + lane) << 4)];
        v16h b0 = *(const v16h*)&ldsB[buf][(((ng * 4 + 0) * 32 + lane) << 4)];
        v16h b1 = *(const v16h*)&ldsB[buf][(((ng * 4 + 1) * 32 + lane) << 4)];
        v16h b2 = *(const v16h*)&ldsB[buf][(((ng * 4 + 2) * 32 + lane) << 4)];
        v16h b3 = *(const v16h*)&ldsB[buf][(((ng * 4 + 3) * 32 + lane) << 4)];
        acc[0][0] = wmma_f16(a0, b0, acc[0][0]);
        acc[1][0] = wmma_f16(a1, b0, acc[1][0]);
        acc[0][1] = wmma_f16(a0, b1, acc[0][1]);
        acc[1][1] = wmma_f16(a1, b1, acc[1][1]);
        acc[0][2] = wmma_f16(a0, b2, acc[0][2]);
        acc[1][2] = wmma_f16(a1, b2, acc[1][2]);
        acc[0][3] = wmma_f16(a0, b3, acc[0][3]);
        acc[1][3] = wmma_f16(a1, b3, acc[1][3]);
    };

    stage(0, 0);
    cp_wait();
    __syncthreads();
    for (int kc = 0; kc < KC; ++kc) {
        int cur = kc & 1;
        if (kc + 1 < KC) stage(cur ^ 1, kc + 1);
        compute(cur);
        cp_wait();
        __syncthreads();
    }

    const int nl = lane & 15, half = lane >> 4;
#pragma unroll
    for (int sm = 0; sm < 2; ++sm)
#pragma unroll
        for (int nt = 0; nt < 4; ++nt) {
            int n = n0 + (ng * 4 + nt) * 16 + nl;
            float bb = bias[n];
#pragma unroll
            for (int v = 0; v < 8; ++v) {
                int m = m0 + (mt * 2 + sm) * 16 + v + half * 8;
                float z = acc[sm][nt][v] + bb;
                z = z > 0.f ? z : expm1f(z);
                Z[(size_t)m * DFF + n] = (_Float16)z;
            }
        }
}

// ---------------- Output GEMV: wave-per-row, vectorized f16 reads -------------------
__global__ __launch_bounds__(256)
void out_gemv(const _Float16* __restrict__ Z2, const float* __restrict__ Wout,
              const float* __restrict__ bout, float* __restrict__ out, int M)
{
    int row = blockIdx.x * 8 + (threadIdx.x >> 5);
    int lane = threadIdx.x & 31;
    if (row >= M) return;
    const _Float16* z = Z2 + (size_t)row * DFF;
    float s = 0.f;
#pragma unroll
    for (int kb = 0; kb < 8; ++kb) {
        int k = kb * 256 + lane * 8;
        Pack8 p;
        p.u = *(const uint4*)(z + k);
#pragma unroll
        for (int e = 0; e < 8; ++e) s += (float)p.h[e] * Wout[k + e];
    }
#pragma unroll
    for (int off = 16; off > 0; off >>= 1) s += __shfl_xor(s, off, 32);
    if (lane == 0) out[row] = s + bout[0];
}

// ---------------- Weight prep: emit fragment-swizzled layouts -----------------------
// Wsw[((kc*(N/16)+nt)*32 + lane)*16 + slot] ; kk = (slot&7) + ((lane>>4)<<3) + ((slot>>3)<<4)
__global__ void build_gates_Wsw(const float* __restrict__ Wih,
                                const float* __restrict__ Whh, int in_dim,
                                _Float16* __restrict__ Wsw)
{
    int KC = (in_dim + DM) >> 5;
    size_t total = (size_t)KC * (NG >> 4) * 512;
    for (size_t i = (size_t)blockIdx.x * blockDim.x + threadIdx.x; i < total;
         i += (size_t)gridDim.x * blockDim.x) {
        int slot = (int)(i & 15);
        int lane = (int)((i >> 4) & 31);
        size_t t = i >> 9;
        int nt = (int)(t % (NG >> 4));
        int kc = (int)(t / (NG >> 4));
        int kk = (slot & 7) + ((lane >> 4) << 3) + ((slot >> 3) << 4);
        int k = kc * 32 + kk;
        int n = nt * 16 + (lane & 15);
        float v = (k < in_dim) ? Wih[(size_t)n * in_dim + k]
                               : Whh[(size_t)n * DM + (k - in_dim)];
        Wsw[i] = (_Float16)v;
    }
}

__global__ void build_Wsw(const float* __restrict__ Wsrc, int K, int N,
                          _Float16* __restrict__ Wsw)
{
    size_t total = (size_t)(K >> 5) * (N >> 4) * 512;
    for (size_t i = (size_t)blockIdx.x * blockDim.x + threadIdx.x; i < total;
         i += (size_t)gridDim.x * blockDim.x) {
        int slot = (int)(i & 15);
        int lane = (int)((i >> 4) & 31);
        size_t t = i >> 9;
        int nt = (int)(t % (N >> 4));
        int kc = (int)(t / (N >> 4));
        int kk = (slot & 7) + ((lane >> 4) << 3) + ((slot >> 3) << 4);
        int k = kc * 32 + kk;
        int n = nt * 16 + (lane & 15);
        Wsw[i] = (_Float16)Wsrc[(size_t)k * N + n];
    }
}

__global__ void sum_bias(const float* __restrict__ a, const float* __restrict__ b,
                         float* __restrict__ o, int n)
{
    int i = blockIdx.x * blockDim.x + threadIdx.x;
    if (i < n) o[i] = a[i] + b[i];
}

// ---------------- Host orchestration ------------------------------------------------
extern "C" void kernel_launch(void* const* d_in, const int* in_sizes, int n_in,
                              void* d_out, int out_size, void* d_ws, size_t ws_size,
                              hipStream_t stream)
{
    const float* y       = (const float*)d_in[0];
    const float* y_draws = (const float*)d_in[1];
    const float* Wih0 = (const float*)d_in[2];
    const float* Whh0 = (const float*)d_in[3];
    const float* bih0 = (const float*)d_in[4];
    const float* bhh0 = (const float*)d_in[5];
    const float* Wih1 = (const float*)d_in[6];
    const float* Whh1 = (const float*)d_in[7];
    const float* bih1 = (const float*)d_in[8];
    const float* bhh1 = (const float*)d_in[9];
    const float* W0   = (const float*)d_in[10];
    const float* b0   = (const float*)d_in[11];
    const float* W1   = (const float*)d_in[12];
    const float* b1   = (const float*)d_in[13];
    const float* Wout = (const float*)d_in[14];
    const float* bout = (const float*)d_in[15];
    float* out = (float*)d_out;

    char* ws = (char*)d_ws;
    size_t off = 0;
    auto alloc = [&](size_t bytes) -> void* {
        void* p = ws + off;
        off += bytes;
        off = (off + 255) & ~(size_t)255;
        return p;
    };
    _Float16* Wg0sw = (_Float16*)alloc((size_t)(32 + DM) * NG * 2);
    _Float16* Wg1sw = (_Float16*)alloc((size_t)(DM + DM) * NG * 2);
    float*    bias0 = (float*)alloc(NG * 4);
    float*    bias1 = (float*)alloc(NG * 4);
    _Float16* W0sw  = (_Float16*)alloc((size_t)KIN * DFF * 2);
    _Float16* W1sw  = (_Float16*)alloc((size_t)DFF * DFF * 2);
    float* hst[8];
    for (int i = 0; i < 8; ++i) hst[i] = (float*)alloc((size_t)BB * DM * 4);
    float** h0 = &hst[0]; float** c0 = &hst[2];   // [parity]
    float** h1 = &hst[4]; float** c1 = &hst[6];
    _Float16* wh16 = (_Float16*)alloc((size_t)BB * TT * DM * 2);
    _Float16* wc16 = (_Float16*)alloc((size_t)BB * TT * DM * 2);
    _Float16* Z1   = (_Float16*)alloc((size_t)MROWS * DFF * 2);
    _Float16* Z2   = (_Float16*)alloc((size_t)MROWS * DFF * 2);
    (void)ws_size; (void)n_in; (void)in_sizes; (void)out_size;

    // zero init: LSTM states + lagged-state buffers (t=0 slot must be zero)
    for (int i = 0; i < 8; ++i)
        (void)hipMemsetAsync(hst[i], 0, (size_t)BB * DM * 4, stream);
    (void)hipMemsetAsync(wh16, 0, (size_t)BB * TT * DM * 2, stream);
    (void)hipMemsetAsync(wc16, 0, (size_t)BB * TT * DM * 2, stream);

    // weight prep: f16, fragment-swizzled (gate weights also transposed + fused)
    build_gates_Wsw<<<2048, 256, 0, stream>>>(Wih0, Whh0, 32, Wg0sw);
    build_gates_Wsw<<<2048, 256, 0, stream>>>(Wih1, Whh1, DM, Wg1sw);
    sum_bias<<<(NG + 255) / 256, 256, 0, stream>>>(bih0, bhh0, bias0, NG);
    sum_bias<<<(NG + 255) / 256, 256, 0, stream>>>(bih1, bhh1, bias1, NG);
    build_Wsw<<<4096, 256, 0, stream>>>(W0, KIN, DFF, W0sw);
    build_Wsw<<<4096, 256, 0, stream>>>(W1, DFF, DFF, W1sw);

    // LSTM recurrence: 2 kernels per timestep, ping-pong states
    for (int t = 0; t < TT; ++t) {
        int rp = t & 1, wp = rp ^ 1;
        lstm_step<<<32, 256, 0, stream>>>(y, TT * 32, t * 32, 32,
                                          h0[rp], c0[rp], Wg0sw, bias0,
                                          h0[wp], c0[wp], nullptr, nullptr, 0);
        lstm_step<<<32, 256, 0, stream>>>(h0[wp], DM, 0, DM,
                                          h1[rp], c1[rp], Wg1sw, bias1,
                                          h1[wp], c1[wp], wh16, wc16, t + 1);
    }

    // MLP: 5 streams (outs, then 4 draws); PRED_LEN == T so all rows map to output
    dim3 gg(MROWS / 64, DFF / 256);
    for (int s = 0; s < 5; ++s) {
        const float* ysrc = (s == 0) ? y : (y_draws + (size_t)(s - 1) * BB * TT * 32);
        mlp_gemm0<<<gg, 256, 0, stream>>>(ysrc, wh16, wc16, W0sw, b0, Z1);
        mlp_gemm1<<<gg, 256, 0, stream>>>(Z1, W1sw, b1, Z2);
        out_gemv<<<MROWS / 8, 256, 0, stream>>>(Z2, Wout, bout, out + (size_t)s * MROWS,
                                                MROWS);
    }
}
